// Attention_79233556677129
// MI455X (gfx1250) — matrix operand order repaired and checked by reading, not compiled, not run on hardware
//
#include <hip/hip_runtime.h>
#include <math.h>

typedef __bf16 bf16_t;
typedef __attribute__((ext_vector_type(16))) __bf16 bf16x16;
typedef __attribute__((ext_vector_type(8)))  float   v8f;

#define B_   4
#define S_   2048
#define H_   1024
#define NH_  16
#define HD_  64
#define M_   (B_ * S_)        // 8192 rows
#define O3_  (3 * NH_ * HD_)  // 3072 qkv features

static __device__ __forceinline__ v8f vzero() {
    v8f z = {0.f, 0.f, 0.f, 0.f, 0.f, 0.f, 0.f, 0.f};
    return z;
}

static __device__ __forceinline__ v8f wmma_bf16(bf16x16 a, bf16x16 b, v8f c) {
    // D = A(16x32 bf16) x B(32x16 bf16) + C(16x16 f32)
    return __builtin_amdgcn_wmma_f32_16x16x32_bf16(
        false, a, false, b, (short)0, c, false, false);
}

// A-matrix fragment: 16x32 (MxK) bf16, row-major source with leading dim `ld`.
// ISA layout: lane m = lane&15; lanes 0-15 hold K={0..7,16..23}, lanes 16-31 K={8..15,24..31}.
static __device__ __forceinline__ bf16x16 load_a_frag(const bf16_t* base, int ld) {
    int lane = threadIdx.x & 31;
    const bf16_t* p = base + (size_t)(lane & 15) * ld + ((lane & 16) ? 8 : 0);
    union { uint4 u[2]; bf16x16 v; } t;
    t.u[0] = *(const uint4*)(p);        // K = kb .. kb+7
    t.u[1] = *(const uint4*)(p + 16);   // K = kb+16 .. kb+23
    return t.v;
}

// B-matrix fragment: 32x16 (KxN) where memory holds B^T row-major (N rows of K),
// leading dim `ld`. Lane n = lane&15; lanes 0-15 hold K={0..7,16..23}, lanes 16-31 K={8..15,24..31}.
static __device__ __forceinline__ bf16x16 load_bT_frag(const bf16_t* base, int ld) {
    int lane = threadIdx.x & 31;
    const bf16_t* p = base + (size_t)(lane & 15) * ld + ((lane & 16) ? 8 : 0);
    union { uint4 u[2]; bf16x16 v; } t;
    t.u[0] = *(const uint4*)(p);        // K = kb .. kb+7
    t.u[1] = *(const uint4*)(p + 16);   // K = kb+16 .. kb+23
    return t.v;
}

// ---------------------------------------------------------------- fp32 -> bf16
__global__ void cvt_f32_bf16(const float* __restrict__ src, bf16_t* __restrict__ dst, int n) {
    int i = blockIdx.x * blockDim.x + threadIdx.x;
    int stride = gridDim.x * blockDim.x;
    for (; i < n; i += stride) dst[i] = (bf16_t)src[i];
}

// ------------------------------------------------- QKV projection + RoPE split
// Each wave: 32 seq rows x 64 dims (one head of q, k, or v); B-frags reused x2.
__global__ void qkv_rope(const bf16_t* __restrict__ Hb,   // (8192, 1024)
                         const bf16_t* __restrict__ Wq,   // (3072, 1024)
                         const float*  __restrict__ cosT, // (2048, 64)
                         const float*  __restrict__ sinT,
                         bf16_t* __restrict__ Q,          // [b][h][s][d]
                         bf16_t* __restrict__ Kd,         // [b][h][s][d]
                         bf16_t* __restrict__ Vt) {       // [b][h][d][s]
    int wid   = blockIdx.x * (blockDim.x >> 5) + (threadIdx.x >> 5);
    int mtile = wid / 48;          // 256 tiles of 32 rows
    int unit  = wid % 48;          // 48 = 3 (q/k/v) * 16 heads
    int lane  = threadIdx.x & 31;
    int o0    = unit * 64;

    v8f c[2][4];
#pragma unroll
    for (int g = 0; g < 2; ++g)
#pragma unroll
        for (int t = 0; t < 4; ++t) c[g][t] = vzero();

    const bf16_t* Abase = Hb + (size_t)mtile * 32 * H_;
#pragma unroll 2
    for (int k0 = 0; k0 < H_; k0 += 32) {
        bf16x16 a0 = load_a_frag(Abase + k0, H_);
        bf16x16 a1 = load_a_frag(Abase + (size_t)16 * H_ + k0, H_);
#pragma unroll
        for (int t = 0; t < 4; ++t) {
            bf16x16 b = load_bT_frag(Wq + (size_t)(o0 + t * 16) * H_ + k0, H_);
            c[0][t] = wmma_bf16(a0, b, c[0][t]);
            c[1][t] = wmma_bf16(a1, b, c[1][t]);
        }
    }

    int which = unit >> 4;                 // 0=q 1=k 2=v
    int head  = unit & 15;
    int n     = lane & 15;
    int mbase = (lane & 16) ? 8 : 0;
    int gm0   = mtile * 32;
    int bidx  = gm0 / S_;
    int srow0 = gm0 % S_;

    if (which == 2) {
        // V stored transposed [b][h][d][s] so attention B-fragments are contiguous
#pragma unroll
        for (int g = 0; g < 2; ++g)
#pragma unroll
            for (int t = 0; t < 4; ++t) {
                int d = t * 16 + n;
                bf16_t* vp = Vt + (((size_t)bidx * NH_ + head) * HD_ + d) * S_;
#pragma unroll
                for (int r = 0; r < 8; ++r)
                    vp[srow0 + g * 16 + mbase + r] = (bf16_t)c[g][t][r];
            }
    } else {
        bf16_t* out = (which == 0 ? Q : Kd) + ((size_t)bidx * NH_ + head) * S_ * HD_;
        // rotate_half partner of (t, n) is (t^2, n): same lane, same VGPR slot
#pragma unroll
        for (int g = 0; g < 2; ++g)
#pragma unroll
            for (int t = 0; t < 4; ++t) {
                int d = t * 16 + n;
#pragma unroll
                for (int r = 0; r < 8; ++r) {
                    int s = srow0 + g * 16 + mbase + r;
                    float cs  = cosT[s * HD_ + d];
                    float sn  = sinT[s * HD_ + d];
                    float x   = c[g][t][r];
                    float rot = (t < 2) ? -c[g][t + 2][r] : c[g][t - 2][r];
                    out[(size_t)s * HD_ + d] = (bf16_t)(x * cs + rot * sn);
                }
            }
    }
}

// ----------------------------------------------------------- flash attention
// One wave per (batch, head, 32-row q tile); online softmax over 32-key chunks.
// K and V fragments are each reused by both 16-row q halves.
__global__ void __launch_bounds__(32) flash_attn(const bf16_t* __restrict__ Q,
                                                 const bf16_t* __restrict__ Kd,
                                                 const bf16_t* __restrict__ Vt,
                                                 bf16_t* __restrict__ Attn) { // (8192, 1024)
    __shared__ __align__(16) bf16_t Plds[32 * 32];

    int bid  = blockIdx.x;
    int qt   = bid & 63;           // 2048/32 q tiles
    int h    = (bid >> 6) & 15;
    int b    = bid >> 10;
    int lane = threadIdx.x & 31;
    int n     = lane & 15;
    int mbase = (lane & 16) ? 8 : 0;

    const bf16_t* Qb = Q  + (((size_t)b * NH_ + h) * S_ + qt * 32) * HD_;
    const bf16_t* Kb = Kd + ((size_t)b * NH_ + h) * S_ * HD_;
    const bf16_t* Vb = Vt + ((size_t)b * NH_ + h) * HD_ * S_;

    bf16x16 aq[2][2];
#pragma unroll
    for (int g = 0; g < 2; ++g) {
        aq[g][0] = load_a_frag(Qb + (size_t)g * 16 * HD_ + 0,  HD_);
        aq[g][1] = load_a_frag(Qb + (size_t)g * 16 * HD_ + 32, HD_);
    }

    float mrow[2][8], lrow[2][8];
#pragma unroll
    for (int g = 0; g < 2; ++g)
#pragma unroll
        for (int r = 0; r < 8; ++r) { mrow[g][r] = -3.0e38f; lrow[g][r] = 0.f; }
    v8f o[2][4];
#pragma unroll
    for (int g = 0; g < 2; ++g)
#pragma unroll
        for (int t = 0; t < 4; ++t) o[g][t] = vzero();

    const float scale = 0.125f;    // 1/sqrt(64)

    for (int s0 = 0; s0 < S_; s0 += 32) {
        if (s0 + 32 < S_)
            __builtin_prefetch(Kb + (size_t)(s0 + 32) * HD_, 0, 0);

        // shared K fragments for this 32-key chunk: [key subtile][k-step]
        bf16x16 bk[2][2];
#pragma unroll
        for (int kt = 0; kt < 2; ++kt) {
            bk[kt][0] = load_bT_frag(Kb + (size_t)(s0 + kt * 16) * HD_ + 0,  HD_);
            bk[kt][1] = load_bT_frag(Kb + (size_t)(s0 + kt * 16) * HD_ + 32, HD_);
        }

        // scores: Q(32x64) . K^T(64x32) -> 2x2 16x16 f32 frags
        v8f sc[2][2];
#pragma unroll
        for (int g = 0; g < 2; ++g)
#pragma unroll
            for (int kt = 0; kt < 2; ++kt) {
                v8f z = vzero();
                z = wmma_bf16(aq[g][0], bk[kt][0], z);
                z = wmma_bf16(aq[g][1], bk[kt][1], z);
                sc[g][kt] = z;
            }

        // online softmax: shfl_xor 1/2/4/8 stays within each 16-lane half,
        // matching C-layout row ownership (lanes<16: M=0..7, lanes>=16: M=8..15)
#pragma unroll
        for (int g = 0; g < 2; ++g) {
            float alpha[8];
#pragma unroll
            for (int r = 0; r < 8; ++r) {
                float s0v = sc[g][0][r] * scale;
                float s1v = sc[g][1][r] * scale;
                float mx = fmaxf(s0v, s1v);
#pragma unroll
                for (int off = 1; off < 16; off <<= 1)
                    mx = fmaxf(mx, __shfl_xor(mx, off, 32));
                float mnew = fmaxf(mrow[g][r], mx);
                alpha[r] = __expf(mrow[g][r] - mnew);
                mrow[g][r] = mnew;
                float p0 = __expf(s0v - mnew);
                float p1 = __expf(s1v - mnew);
                sc[g][0][r] = p0; sc[g][1][r] = p1;
                float rs = p0 + p1;
#pragma unroll
                for (int off = 1; off < 16; off <<= 1)
                    rs += __shfl_xor(rs, off, 32);
                lrow[g][r] = lrow[g][r] * alpha[r] + rs;
            }
#pragma unroll
            for (int t = 0; t < 4; ++t)
#pragma unroll
                for (int r = 0; r < 8; ++r)
                    o[g][t][r] *= alpha[r];

            // C-layout -> A-layout transpose of P through LDS (row g*16+m, col key)
#pragma unroll
            for (int kt = 0; kt < 2; ++kt)
#pragma unroll
                for (int r = 0; r < 8; ++r)
                    Plds[(g * 16 + mbase + r) * 32 + kt * 16 + n] = (bf16_t)sc[g][kt][r];
        }
        __syncthreads();
        bf16x16 pa0 = load_a_frag(Plds,           32);
        bf16x16 pa1 = load_a_frag(Plds + 16 * 32, 32);
        __syncthreads();

        // O += P(32x32) . V(32x64); V fragments shared across both q halves
#pragma unroll
        for (int t = 0; t < 4; ++t) {
            bf16x16 bv = load_bT_frag(Vb + (size_t)(t * 16) * S_ + s0, S_);
            o[0][t] = wmma_bf16(pa0, bv, o[0][t]);
            o[1][t] = wmma_bf16(pa1, bv, o[1][t]);
        }
    }

    int srow0 = qt * 32;
#pragma unroll
    for (int g = 0; g < 2; ++g) {
        float inv[8];
#pragma unroll
        for (int r = 0; r < 8; ++r) inv[r] = 1.f / lrow[g][r];
#pragma unroll
        for (int t = 0; t < 4; ++t)
#pragma unroll
            for (int r = 0; r < 8; ++r)
                Attn[((size_t)b * S_ + srow0 + g * 16 + mbase + r) * H_ +
                     h * HD_ + t * 16 + n] = (bf16_t)(o[g][t][r] * inv[r]);
    }
}

// -------------------------------------------------------------- out = attn . w_o^T
// Each wave: 32 rows x 64 cols; B-frags reused x2.
__global__ void out_proj(const bf16_t* __restrict__ Attn, // (8192, 1024)
                         const bf16_t* __restrict__ Wo,   // (1024, 1024)
                         float* __restrict__ Out) {       // (8192, 1024) f32
    int wid   = blockIdx.x * (blockDim.x >> 5) + (threadIdx.x >> 5);
    int mtile = wid >> 4;          // 256 tiles of 32 rows
    int nunit = wid & 15;          // 16 units of 64 cols
    int lane  = threadIdx.x & 31;

    v8f c[2][4];
#pragma unroll
    for (int g = 0; g < 2; ++g)
#pragma unroll
        for (int t = 0; t < 4; ++t) c[g][t] = vzero();

    const bf16_t* Abase = Attn + (size_t)mtile * 32 * H_;
#pragma unroll 2
    for (int k0 = 0; k0 < H_; k0 += 32) {
        bf16x16 a0 = load_a_frag(Abase + k0, H_);
        bf16x16 a1 = load_a_frag(Abase + (size_t)16 * H_ + k0, H_);
#pragma unroll
        for (int t = 0; t < 4; ++t) {
            bf16x16 b = load_bT_frag(Wo + (size_t)(nunit * 64 + t * 16) * H_ + k0, H_);
            c[0][t] = wmma_bf16(a0, b, c[0][t]);
            c[1][t] = wmma_bf16(a1, b, c[1][t]);
        }
    }

    int n     = lane & 15;
    int mbase = (lane & 16) ? 8 : 0;
#pragma unroll
    for (int g = 0; g < 2; ++g)
#pragma unroll
        for (int t = 0; t < 4; ++t)
#pragma unroll
            for (int r = 0; r < 8; ++r)
                Out[((size_t)mtile * 32 + g * 16 + mbase + r) * H_ +
                    nunit * 64 + t * 16 + n] = c[g][t][r];
}

extern "C" void kernel_launch(void* const* d_in, const int* in_sizes, int n_in,
                              void* d_out, int out_size, void* d_ws, size_t ws_size,
                              hipStream_t stream) {
    (void)in_sizes; (void)n_in; (void)out_size; (void)ws_size;
    const float* cosT   = (const float*)d_in[0];
    const float* sinT   = (const float*)d_in[1];
    const float* hidden = (const float*)d_in[2];
    const float* w_qkv  = (const float*)d_in[3];
    const float* w_o    = (const float*)d_in[4];
    float* out = (float*)d_out;

    bf16_t* ws = (bf16_t*)d_ws;
    size_t off = 0;
    bf16_t* Hb = ws + off; off += (size_t)M_ * H_;    // hidden bf16
    bf16_t* Wq = ws + off; off += (size_t)O3_ * H_;   // w_qkv bf16
    bf16_t* Wo = ws + off; off += (size_t)H_ * H_;    // w_o bf16
    bf16_t* Q  = ws + off; off += (size_t)M_ * H_;    // [b][h][s][d]
    bf16_t* Kd = ws + off; off += (size_t)M_ * H_;    // [b][h][s][d]
    bf16_t* Vt = ws + off; off += (size_t)M_ * H_;    // [b][h][d][s]
    bf16_t* At = ws + off; off += (size_t)M_ * H_;    // attn pre-projection

    cvt_f32_bf16<<<4096, 256, 0, stream>>>(hidden, Hb, M_ * H_);
    cvt_f32_bf16<<<2048, 256, 0, stream>>>(w_qkv, Wq, O3_ * H_);
    cvt_f32_bf16<<<1024, 256, 0, stream>>>(w_o, Wo, H_ * H_);

    // 256 m-tiles * 48 (3 * 16 heads) waves, 8 waves/block
    qkv_rope<<<1536, 256, 0, stream>>>(Hb, Wq, cosT, sinT, Q, Kd, Vt);

    // 4 * 16 * 64 = 4096 (b, head, 32-row q-tile) waves, one per block
    flash_attn<<<4096, 32, 0, stream>>>(Q, Kd, Vt, At);

    // 256 m-tiles * 16 n-units, 8 waves/block
    out_proj<<<512, 256, 0, stream>>>(At, Wo, out);
}
